// LinearChainCRF_22548578304233
// MI455X (gfx1250) — matrix operand order repaired and checked by reading, not compiled, not run on hardware
//
#include <hip/hip_runtime.h>

// Linear-chain CRF forward (log-partition) for B=64, T=4096, S=64.
// Scaled-exp-domain scan: q_t[j] = exp(alpha_t[j] - L_t), max q = 1 exactly.
// Each step: one 16x64 @ 64x64 GEMM as D^T = E^T @ P^T using
// v_wmma_f32_16x16x32_f16, then w = acc*exp(s), mu = max w, q' = w/mu,
// L' = L + maxT + ln(mu). One v_log_f32 per step; the 32 exp2(s) co-execute
// with the WMMAs. The j-permutation j(n,g,k) = k + 8(n&1) + 32(n>>1) + 16g
// is baked into the constant E^T fragments so the WMMA D-layout IS the next
// step's B-fragment layout: repacking is pure v_cvt_pk_rtz, no shuffles.
// One wave == 16 batch rows, fully independent: no barriers, no LDS traffic
// except a single ds_swizzle xor-16 in the column-max reduction.

#define BATCHN 64
#define SEQN   4096
#define SN     64

typedef __attribute__((ext_vector_type(16))) _Float16 v16h;
typedef __attribute__((ext_vector_type(8)))  float    v8f;

union FragU { v16h h; unsigned int u[8]; };

#define LOG2E_F 1.4426950408889634f
#define LN2_F   0.6931471805599453f

__device__ __forceinline__ float swz16_f(float x) {
  return __int_as_float(__builtin_amdgcn_ds_swizzle(__float_as_int(x), 0x401F)); // xor 16
}

// max over the lane's 32 values (pairwise tree, depth 5) + partner half-group
__device__ __forceinline__ float max32_tree(const float (&w)[4][8]) {
  float m16[16];
#pragma unroll
  for (int n = 0; n < 4; ++n)
#pragma unroll
    for (int k = 0; k < 4; ++k)
      m16[n * 4 + k] = fmaxf(w[n][k], w[n][k + 4]);
  float m8[8];
#pragma unroll
  for (int i = 0; i < 8; ++i) m8[i] = fmaxf(m16[i], m16[i + 8]);
  float m4[4];
#pragma unroll
  for (int i = 0; i < 4; ++i) m4[i] = fmaxf(m8[i], m8[i + 4]);
  float m = fmaxf(fmaxf(m4[0], m4[2]), fmaxf(m4[1], m4[3]));
  return fmaxf(m, swz16_f(m)); // combine with partner half-group -> column max
}

// D-layout == B-layout (thanks to the j-permutation): pure pack, no shuffles.
__device__ __forceinline__ void pack_frags(const float (&w)[4][8], float rq,
                                           FragU &b0, FragU &b1) {
#pragma unroll
  for (int k = 0; k < 4; ++k) {
    b0.u[k]     = __builtin_bit_cast(unsigned int,
                    __builtin_amdgcn_cvt_pkrtz(w[0][2 * k] * rq, w[0][2 * k + 1] * rq));
    b0.u[4 + k] = __builtin_bit_cast(unsigned int,
                    __builtin_amdgcn_cvt_pkrtz(w[1][2 * k] * rq, w[1][2 * k + 1] * rq));
    b1.u[k]     = __builtin_bit_cast(unsigned int,
                    __builtin_amdgcn_cvt_pkrtz(w[2][2 * k] * rq, w[2][2 * k + 1] * rq));
    b1.u[4 + k] = __builtin_bit_cast(unsigned int,
                    __builtin_amdgcn_cvt_pkrtz(w[3][2 * k] * rq, w[3][2 * k + 1] * rq));
  }
}

__global__ __launch_bounds__(32) void crf_fwd_kernel(
    const float* __restrict__ scores, const float* __restrict__ trans,
    const float* __restrict__ source, const float* __restrict__ sink,
    float* __restrict__ out) {
  const int lane = threadIdx.x & 31;
  const int r = lane & 15;   // batch column within the 16-wide tile
  const int g = lane >> 4;   // half-group (0: lanes 0-15, 1: lanes 16-31)
  const long b = (long)blockIdx.x * 16 + r;

  // tile n, element k of this lane covers state j = jbase[n] + k
  int jbase[4];
#pragma unroll
  for (int n = 0; n < 4; ++n) jbase[n] = 8 * (n & 1) + 32 * (n >> 1) + 16 * g;

  // ---- global max of transition (normalizer for f16 exp-space E) ----
  float mt = -3.4e38f;
#pragma unroll 8
  for (int q = 0; q < (SN * SN) / 32; ++q)
    mt = fmaxf(mt, trans[lane + 32 * q]);
  mt = fmaxf(mt, __int_as_float(__builtin_amdgcn_ds_swizzle(__float_as_int(mt), 0x041F)));
  mt = fmaxf(mt, __int_as_float(__builtin_amdgcn_ds_swizzle(__float_as_int(mt), 0x081F)));
  mt = fmaxf(mt, __int_as_float(__builtin_amdgcn_ds_swizzle(__float_as_int(mt), 0x101F)));
  mt = fmaxf(mt, __int_as_float(__builtin_amdgcn_ds_swizzle(__float_as_int(mt), 0x201F)));
  mt = fmaxf(mt, swz16_f(mt));

  // ---- constant A-fragments of E^T with the permuted row mapping ----
  // A tile n, row M corresponds to j = (M&7) + 8(n&1) + 32(n>>1) + 16(M>>3).
  // 16-bit A 16x32 layout (contiguous, matches B): lanes<16: K = h; lanes>=16: +16.
  FragU afrag[4][2];
#pragma unroll
  for (int n = 0; n < 4; ++n) {
    const int j = (r & 7) + 8 * (n & 1) + 32 * (n >> 1) + 16 * (r >> 3);
#pragma unroll
    for (int f = 0; f < 2; ++f) {
#pragma unroll
      for (int h2i = 0; h2i < 8; ++h2i) {
        const int h0 = 2 * h2i, h1 = 2 * h2i + 1;
        const int K0 = h0 + 16 * g + 32 * f; // i index
        const int K1 = h1 + 16 * g + 32 * f;
        float e0 = __builtin_amdgcn_exp2f((trans[K0 * SN + j] - mt) * LOG2E_F);
        float e1 = __builtin_amdgcn_exp2f((trans[K1 * SN + j] - mt) * LOG2E_F);
        afrag[n][f].u[h2i] = __builtin_bit_cast(unsigned int,
                               __builtin_amdgcn_cvt_pkrtz(e0, e1));
      }
    }
  }

  // ---- t = 0: w = exp(source + scores[:,0] + log S); L = ln(max w) ----
  const float* sbase = scores + (size_t)b * SEQN * SN;
  float w[4][8]; // lane holds scaled exp-alpha for j = jbase[n] + k
  const float LOG64 = 4.1588830833596715f;
#pragma unroll
  for (int n = 0; n < 4; ++n) {
    const int jb = jbase[n];
    float4 sa = *(const float4*)(sbase + jb);
    float4 sb = *(const float4*)(sbase + jb + 4);
    float4 qa = *(const float4*)(source + jb);
    float4 qb = *(const float4*)(source + jb + 4);
    const float av[8] = {sa.x + qa.x, sa.y + qa.y, sa.z + qa.z, sa.w + qa.w,
                         sb.x + qb.x, sb.y + qb.y, sb.z + qb.z, sb.w + qb.w};
#pragma unroll
    for (int k = 0; k < 8; ++k)
      w[n][k] = __builtin_amdgcn_exp2f((av[k] + LOG64) * LOG2E_F);
  }
  float mu = max32_tree(w);
  float rq = __builtin_amdgcn_rcpf(mu);
  float L = __builtin_amdgcn_logf(mu) * LN2_F;

  FragU bf0, bf1;
  pack_frags(w, rq, bf0, bf1);

  // scores register pipeline (one step ahead)
  float4 sc[4][2];
#pragma unroll
  for (int n = 0; n < 4; ++n) {
    const float* p = sbase + (size_t)1 * SN + jbase[n];
    sc[n][0] = *(const float4*)p; sc[n][1] = *(const float4*)(p + 4);
  }

  for (int t = 1; t < SEQN; ++t) {
    // far prefetch: pull this lane's batch row 8 steps ahead into cache
    {
      const int tp = (t + 8 < SEQN) ? (t + 8) : (SEQN - 1);
      __builtin_prefetch(sbase + (size_t)tp * SN + 32 * g, 0, 3);
    }
    // near prefetch into registers: step t+1
    float4 snx[4][2];
    const int tn = (t + 1 < SEQN) ? (t + 1) : t;
#pragma unroll
    for (int n = 0; n < 4; ++n) {
      const float* p = sbase + (size_t)tn * SN + jbase[n];
      snx[n][0] = *(const float4*)p; snx[n][1] = *(const float4*)(p + 4);
    }

#pragma unroll
    for (int n = 0; n < 4; ++n) {
      // exp(scores): independent of the WMMA result -> co-executes with it
      const float sv[8] = {sc[n][0].x, sc[n][0].y, sc[n][0].z, sc[n][0].w,
                           sc[n][1].x, sc[n][1].y, sc[n][1].z, sc[n][1].w};
      float es[8];
#pragma unroll
      for (int k = 0; k < 8; ++k)
        es[k] = __builtin_amdgcn_exp2f(sv[k] * LOG2E_F);

      v8f c = {};
      c = __builtin_amdgcn_wmma_f32_16x16x32_f16(false, afrag[n][0].h,
                                                 false, bf0.h, (short)0, c,
                                                 false, false);
      c = __builtin_amdgcn_wmma_f32_16x16x32_f16(false, afrag[n][1].h,
                                                 false, bf1.h, (short)0, c,
                                                 false, false);
#pragma unroll
      for (int k = 0; k < 8; ++k) w[n][k] = c[k] * es[k];
    }

    mu = max32_tree(w);                  // > 0 always (prev q_max == 1 in f16)
    rq = __builtin_amdgcn_rcpf(mu);
    L  = __builtin_fmaf(__builtin_amdgcn_logf(mu), LN2_F, L + mt);
    pack_frags(w, rq, bf0, bf1);

#pragma unroll
    for (int n = 0; n < 4; ++n) { sc[n][0] = snx[n][0]; sc[n][1] = snx[n][1]; }
  }

  // ---- final: out[b] = L + ln( sum_j q_j * exp(sink_j) ) ----
  float ssum = 0.f;
#pragma unroll
  for (int n = 0; n < 4; ++n) {
    const int jb = jbase[n];
    float4 ka = *(const float4*)(sink + jb);
    float4 kb = *(const float4*)(sink + jb + 4);
    const float kk[8] = {ka.x, ka.y, ka.z, ka.w, kb.x, kb.y, kb.z, kb.w};
#pragma unroll
    for (int k = 0; k < 8; ++k)
      ssum += (w[n][k] * rq) * __builtin_amdgcn_exp2f(kk[k] * LOG2E_F);
  }
  ssum += swz16_f(ssum);
  if (g == 0)
    out[(long)blockIdx.x * 16 + r] =
        __builtin_fmaf(__builtin_amdgcn_logf(ssum), LN2_F, L);
}

extern "C" void kernel_launch(void* const* d_in, const int* in_sizes, int n_in,
                              void* d_out, int out_size, void* d_ws,
                              size_t ws_size, hipStream_t stream) {
  (void)in_sizes; (void)n_in; (void)out_size; (void)d_ws; (void)ws_size;
  const float* scores = (const float*)d_in[0];
  const float* trans  = (const float*)d_in[1];
  const float* source = (const float*)d_in[2];
  const float* sink   = (const float*)d_in[3];
  float* out = (float*)d_out;
  crf_fwd_kernel<<<dim3(BATCHN / 16), dim3(32), 0, stream>>>(
      scores, trans, source, sink, out);
}
